// JointAttention_28140625723411
// MI455X (gfx1250) — matrix operand order repaired and checked by reading, not compiled, not run on hardware
//
#include <hip/hip_runtime.h>
#include <hip/hip_bf16.h>

// ---------------------------------------------------------------------------
// JointAttention for MI455X (gfx1250): bf16 WMMA everywhere.
//   B=2, Tq=2048, Tk=4096 (self 2048 + context 2048), D=1024, H=16, hd=64
// Pipeline:
//   1) convert x/context fp32->bf16; convert+transpose all weights (N x K)
//   2) WMMA GEMM: Q = x Wq + bq              -> Qb  [B*Tx][D]    bf16
//                 K = [x Wks | ctx Wkc] + b  -> Kb  [B*Tk][D]    bf16
//                 V = [x Wvs | ctx Wvc] + b  -> Vtb [B][D][Tk]   bf16 (transposed)
//   3) flash-attention (online softmax) with WMMA for QK^T and PV -> Ab bf16
//   4) WMMA GEMM: out = Ab Wo + bo -> d_out fp32
// ---------------------------------------------------------------------------

typedef __attribute__((ext_vector_type(16))) __bf16 bf16x16;
typedef __attribute__((ext_vector_type(8)))  __bf16 bf16x8;
typedef __attribute__((ext_vector_type(8)))  float  f32x8;

#define D_MODEL   1024
#define NUM_HEADS 16
#define HEAD_DIM  64
#define BATCH     2
#define TX        2048
#define TKTOT     4096

static __device__ __forceinline__ bf16x16 cat8(bf16x8 lo, bf16x8 hi) {
  return __builtin_shufflevector(lo, hi, 0,1,2,3,4,5,6,7,8,9,10,11,12,13,14,15);
}

static __device__ __forceinline__ f32x8 wmma_bf16(bf16x16 a, bf16x16 b, f32x8 c) {
  return __builtin_amdgcn_wmma_f32_16x16x32_bf16(false, a, false, b,
                                                 (short)0, c, false, false);
}

// --------------------------- conversion kernels ----------------------------

__global__ void cvt_f32_bf16(const float* __restrict__ in,
                             __bf16* __restrict__ out, int n) {
  for (int i = blockIdx.x * blockDim.x + threadIdx.x; i < n;
       i += gridDim.x * blockDim.x)
    out[i] = (__bf16)in[i];
}

// W [K x N] fp32 -> Wt [N x K] bf16 (so WMMA B-fragments read contiguous K)
__global__ void cvt_transpose_w(const float* __restrict__ W,
                                __bf16* __restrict__ Wt) {
  int idx = blockIdx.x * blockDim.x + threadIdx.x;
  if (idx < D_MODEL * D_MODEL) {
    int k = idx >> 10;          // idx / D_MODEL
    int n = idx & 1023;         // idx % D_MODEL
    Wt[(size_t)n * D_MODEL + k] = (__bf16)W[idx];
  }
}

// ------------------------------ WMMA GEMM ----------------------------------
// C[M x N] = A[M x K] @ W[K x N] + bias, A row-major bf16, Wt = W^T (N x K).
// 8 waves/block: 2 along M (32 rows each), 4 along N (64 cols each).
// Wave tile 32x64: per K-step 12 b128 loads feed 8 WMMAs (1.5 loads/WMMA).
// mode 0: bf16 out, row = b*Tout + toff + t      (Q / K-concat)
// mode 1: bf16 out, transposed V: [b][n][toff+t] (row stride TkT)
// mode 2: fp32 out, row = m                      (final projection)
__global__ __launch_bounds__(256) void gemm_bf16_wmma(
    const __bf16* __restrict__ A, const __bf16* __restrict__ Wt,
    const float* __restrict__ bias, void* __restrict__ out,
    int K, int N, int mode, int Tx, int Tout, int toff, int TkT) {
  const int lane  = threadIdx.x & 31;
  const int wave  = threadIdx.x >> 5;
  const int mbase = blockIdx.x * 64  + (wave & 1) * 32;
  const int nbase = blockIdx.y * 256 + (wave >> 1) * 64;

  const int ar = lane & 15;          // A fragment row
  const int ak = (lane >> 4) * 8;    // A fragment K-chunk base
  const int bn = lane & 15;          // B fragment column
  const int bk = (lane >> 4) * 8;    // B fragment K-chunk base

  f32x8 acc[2][4];
#pragma unroll
  for (int r = 0; r < 2; ++r)
#pragma unroll
    for (int t = 0; t < 4; ++t) acc[r][t] = {};

  const __bf16* Arow0 = A + (size_t)(mbase + ar) * K;
  const __bf16* Arow1 = A + (size_t)(mbase + 16 + ar) * K;
  const __bf16* Wr[4];
#pragma unroll
  for (int t = 0; t < 4; ++t)
    Wr[t] = Wt + (size_t)(nbase + t * 16 + bn) * K + bk;

  for (int kk = 0; kk < K; kk += 32) {
    __builtin_prefetch(Arow0 + kk + 256, 0, 3);   // global_prefetch_b8
    bf16x16 af0 = cat8(*(const bf16x8*)(Arow0 + kk + ak),
                       *(const bf16x8*)(Arow0 + kk + ak + 16));
    bf16x16 af1 = cat8(*(const bf16x8*)(Arow1 + kk + ak),
                       *(const bf16x8*)(Arow1 + kk + ak + 16));
#pragma unroll
    for (int t = 0; t < 4; ++t) {
      bf16x16 bf = cat8(*(const bf16x8*)(Wr[t] + kk),
                        *(const bf16x8*)(Wr[t] + kk + 16));
      acc[0][t] = wmma_bf16(af0, bf, acc[0][t]);
      acc[1][t] = wmma_bf16(af1, bf, acc[1][t]);
    }
  }

  // C layout: row = i + mo (+16 for r=1), col = lane&15
  const int mo = (lane >> 4) * 8;
  const int bb = mbase / Tx;           // 64-row block never crosses batch edge
  const int base_tt = mbase - bb * Tx;
#pragma unroll
  for (int r = 0; r < 2; ++r) {
#pragma unroll
    for (int t = 0; t < 4; ++t) {
#pragma unroll
      for (int i = 0; i < 8; ++i) {
        int mrel = r * 16 + i + mo;
        int n = nbase + t * 16 + (lane & 15);
        float v = acc[r][t][i] + bias[n];
        if (mode == 2) {
          ((float*)out)[(size_t)(mbase + mrel) * N + n] = v;
        } else {
          int tt = base_tt + mrel;
          if (mode == 0)
            ((__bf16*)out)[((size_t)bb * Tout + toff + tt) * N + n] = (__bf16)v;
          else
            ((__bf16*)out)[(size_t)bb * (size_t)N * TkT + (size_t)n * TkT +
                           (toff + tt)] = (__bf16)v;
        }
      }
    }
  }
}

// --------------------------- flash attention -------------------------------
// grid = (Tx/64, B*H); 4 waves/block, each wave: 16 q-rows x hd=64.
__global__ __launch_bounds__(128) void attn_wmma(
    const __bf16* __restrict__ Qb, const __bf16* __restrict__ Kb,
    const __bf16* __restrict__ Vt, __bf16* __restrict__ Ab) {
  __shared__ __bf16 Pl[4][16][32];   // per-wave P staging (C-layout -> A-layout)

  const int lane = threadIdx.x & 31;
  const int wave = threadIdx.x >> 5;
  const int b    = blockIdx.y >> 4;
  const int h    = blockIdx.y & 15;
  const int qloc = blockIdx.x * 64 + wave * 16;

  const int ar = lane & 15, ak = (lane >> 4) * 16;
  const int bn = lane & 15, bk = (lane >> 4) * 16;

  // Q fragments (A operand, 16x64 -> 2 K-steps of 32), loaded once
  const __bf16* Qrow =
      Qb + (size_t)(b * TX + qloc + ar) * D_MODEL + h * HEAD_DIM;
  bf16x16 qf0 = cat8(*(const bf16x8*)(Qrow + ak),
                     *(const bf16x8*)(Qrow + ak + 8));
  bf16x16 qf1 = cat8(*(const bf16x8*)(Qrow + 32 + ak),
                     *(const bf16x8*)(Qrow + 32 + ak + 8));

  f32x8 o[4];
  o[0] = {}; o[1] = {}; o[2] = {}; o[3] = {};
  float mrow[8], lrow[8];
#pragma unroll
  for (int i = 0; i < 8; ++i) { mrow[i] = -1e30f; lrow[i] = 0.0f; }

  const __bf16* Kbase = Kb + (size_t)b * TKTOT * D_MODEL + h * HEAD_DIM + bk;
  const __bf16* Vbase =
      Vt + ((size_t)b * D_MODEL + h * HEAD_DIM + bn) * TKTOT + bk;

  for (int kb = 0; kb < TKTOT; kb += 32) {
    // ---- S = (Q K^T) * 1/sqrt(hd) : two 16x16 tiles over 32 keys ----
    f32x8 s0 = {}, s1 = {};
    {
      const __bf16* kr0 = Kbase + (size_t)(kb + bn) * D_MODEL;
      const __bf16* kr1 = Kbase + (size_t)(kb + 16 + bn) * D_MODEL;
      bf16x16 k00 = cat8(*(const bf16x8*)(kr0),      *(const bf16x8*)(kr0 + 8));
      bf16x16 k01 = cat8(*(const bf16x8*)(kr0 + 32), *(const bf16x8*)(kr0 + 40));
      bf16x16 k10 = cat8(*(const bf16x8*)(kr1),      *(const bf16x8*)(kr1 + 8));
      bf16x16 k11 = cat8(*(const bf16x8*)(kr1 + 32), *(const bf16x8*)(kr1 + 40));
      s0 = wmma_bf16(qf0, k00, s0);
      s0 = wmma_bf16(qf1, k01, s0);
      s1 = wmma_bf16(qf0, k10, s1);
      s1 = wmma_bf16(qf1, k11, s1);
    }

    // ---- online softmax (per-row over 16 lanes; xor<16 stays in group) ----
#pragma unroll
    for (int i = 0; i < 8; ++i) {
      float a0 = s0[i] * 0.125f, a1 = s1[i] * 0.125f;
      float mx = fmaxf(a0, a1);
      mx = fmaxf(mx, __shfl_xor(mx, 1));
      mx = fmaxf(mx, __shfl_xor(mx, 2));
      mx = fmaxf(mx, __shfl_xor(mx, 4));
      mx = fmaxf(mx, __shfl_xor(mx, 8));
      float mnew = fmaxf(mrow[i], mx);
      float corr = __expf(mrow[i] - mnew);
      float p0 = __expf(a0 - mnew);
      float p1 = __expf(a1 - mnew);
      float rs = p0 + p1;
      rs += __shfl_xor(rs, 1);
      rs += __shfl_xor(rs, 2);
      rs += __shfl_xor(rs, 4);
      rs += __shfl_xor(rs, 8);
      lrow[i] = lrow[i] * corr + rs;
      mrow[i] = mnew;
      o[0][i] *= corr; o[1][i] *= corr; o[2][i] *= corr; o[3][i] *= corr;
      int r = i + ((lane >> 4) << 3);           // C-layout row
      Pl[wave][r][lane & 15]        = (__bf16)p0;
      Pl[wave][r][(lane & 15) + 16] = (__bf16)p1;
    }

    // ---- re-read P as an A fragment (16x32) from wave-private LDS ----
    bf16x16 pf = cat8(*(const bf16x8*)&Pl[wave][ar][ak],
                      *(const bf16x8*)&Pl[wave][ar][ak + 8]);

    // ---- O += P @ V : 4 dim-tiles of 16 ----
#pragma unroll
    for (int t = 0; t < 4; ++t) {
      const __bf16* vr = Vbase + (size_t)t * 16 * TKTOT + kb;
      bf16x16 vf = cat8(*(const bf16x8*)(vr), *(const bf16x8*)(vr + 8));
      o[t] = wmma_bf16(pf, vf, o[t]);
    }
  }

  // ---- normalize and write bf16 attention output ----
  const int mo = (lane >> 4) * 8;
#pragma unroll
  for (int i = 0; i < 8; ++i) {
    float inv = 1.0f / lrow[i];
    int qr = b * TX + qloc + i + mo;
#pragma unroll
    for (int t = 0; t < 4; ++t) {
      int col = h * HEAD_DIM + t * 16 + (lane & 15);
      Ab[(size_t)qr * D_MODEL + col] = (__bf16)(o[t][i] * inv);
    }
  }
}

// ------------------------------- launcher ----------------------------------

extern "C" void kernel_launch(void* const* d_in, const int* in_sizes, int n_in,
                              void* d_out, int out_size, void* d_ws,
                              size_t ws_size, hipStream_t stream) {
  (void)in_sizes; (void)n_in; (void)out_size; (void)ws_size;

  const float* x   = (const float*)d_in[0];
  const float* ctx = (const float*)d_in[1];
  const float* Wq  = (const float*)d_in[2];  const float* bq  = (const float*)d_in[3];
  const float* Wks = (const float*)d_in[4];  const float* bks = (const float*)d_in[5];
  const float* Wvs = (const float*)d_in[6];  const float* bvs = (const float*)d_in[7];
  const float* Wkc = (const float*)d_in[8];  const float* bkc = (const float*)d_in[9];
  const float* Wvc = (const float*)d_in[10]; const float* bvc = (const float*)d_in[11];
  const float* Wo  = (const float*)d_in[12]; const float* bo  = (const float*)d_in[13];

  char* ws = (char*)d_ws;
  size_t off = 0;
  auto alloc = [&](size_t bytes) -> void* {
    void* p = ws + off;
    off += (bytes + 255) & ~(size_t)255;
    return p;
  };

  const size_t actB = (size_t)BATCH * TX * D_MODEL * sizeof(__bf16);   // 8 MB
  const size_t wB   = (size_t)D_MODEL * D_MODEL * sizeof(__bf16);      // 2 MB

  __bf16* xb   = (__bf16*)alloc(actB);
  __bf16* cb   = (__bf16*)alloc(actB);
  __bf16* wtq  = (__bf16*)alloc(wB);
  __bf16* wtks = (__bf16*)alloc(wB);
  __bf16* wtvs = (__bf16*)alloc(wB);
  __bf16* wtkc = (__bf16*)alloc(wB);
  __bf16* wtvc = (__bf16*)alloc(wB);
  __bf16* wto  = (__bf16*)alloc(wB);
  __bf16* Qb   = (__bf16*)alloc(actB);
  __bf16* Kb   = (__bf16*)alloc((size_t)BATCH * TKTOT * D_MODEL * sizeof(__bf16));
  __bf16* Vtb  = (__bf16*)alloc((size_t)BATCH * D_MODEL * TKTOT * sizeof(__bf16));
  __bf16* Ab   = (__bf16*)alloc(actB);

  const int nAct = BATCH * TX * D_MODEL;

  cvt_f32_bf16<<<4096, 256, 0, stream>>>(x,   xb, nAct);
  cvt_f32_bf16<<<4096, 256, 0, stream>>>(ctx, cb, nAct);
  cvt_transpose_w<<<4096, 256, 0, stream>>>(Wq,  wtq);
  cvt_transpose_w<<<4096, 256, 0, stream>>>(Wks, wtks);
  cvt_transpose_w<<<4096, 256, 0, stream>>>(Wvs, wtvs);
  cvt_transpose_w<<<4096, 256, 0, stream>>>(Wkc, wtkc);
  cvt_transpose_w<<<4096, 256, 0, stream>>>(Wvc, wtvc);
  cvt_transpose_w<<<4096, 256, 0, stream>>>(Wo,  wto);

  dim3 ggrid((BATCH * TX) / 64, D_MODEL / 256);   // (64, 4), 64x256 block tile
  // Q
  gemm_bf16_wmma<<<ggrid, 256, 0, stream>>>(xb, wtq, bq, Qb,
      D_MODEL, D_MODEL, /*mode=*/0, TX, TX, 0, 0);
  // K (self | context) concat into [B*Tk][D]
  gemm_bf16_wmma<<<ggrid, 256, 0, stream>>>(xb, wtks, bks, Kb,
      D_MODEL, D_MODEL, 0, TX, TKTOT, 0, 0);
  gemm_bf16_wmma<<<ggrid, 256, 0, stream>>>(cb, wtkc, bkc, Kb,
      D_MODEL, D_MODEL, 0, TX, TKTOT, TX, 0);
  // V (self | context), written transposed: [B][D][Tk]
  gemm_bf16_wmma<<<ggrid, 256, 0, stream>>>(xb, wtvs, bvs, Vtb,
      D_MODEL, D_MODEL, /*mode=*/1, TX, 0, 0, TKTOT);
  gemm_bf16_wmma<<<ggrid, 256, 0, stream>>>(cb, wtvc, bvc, Vtb,
      D_MODEL, D_MODEL, 1, TX, 0, TX, TKTOT);

  // Attention
  attn_wmma<<<dim3(TX / 64, BATCH * NUM_HEADS), 128, 0, stream>>>(Qb, Kb, Vtb, Ab);

  // Output projection -> fp32 d_out
  gemm_bf16_wmma<<<ggrid, 256, 0, stream>>>(Ab, wto, bo, d_out,
      D_MODEL, D_MODEL, /*mode=*/2, TX, TX, 0, 0);
}
